// LMANet_32925219291301
// MI455X (gfx1250) — matrix operand order repaired and checked by reading, not compiled, not run on hardware
//
#include <hip/hip_runtime.h>
#include <hip/hip_bf16.h>

typedef __attribute__((ext_vector_type(16))) _Float16 v16h;
typedef __attribute__((ext_vector_type(8)))  _Float16 v8h;
typedef __attribute__((ext_vector_type(8)))  float    v8f;

#define D_    78
#define DI_   156
#define DS_   32
#define DC_   4
#define DTR_  5
#define DBC_  (DTR_ + 2*DS_)   /* 69 */
#define OD_   128
#define NMOL  32768
#define EMOL  131072
#define BMOL  512
#define NPRO  131072
#define EPRO  1048576
#define NPROT 1024
#define EPPI  16384

#define SHUF16(x, y) __builtin_shufflevector((x), (y), 0,1,2,3,4,5,6,7,8,9,10,11,12,13,14,15)

// ---------------------------------------------------------------------------
// WMMA GEMM: C[M,N] = act( A[M,K] * B[K,N] + bias ), f32 in/out, f16 compute,
// f32 accumulate. Block = 4 waves = 32x64 output tile (each wave: two 16x16
// tiles stacked in M sharing one B fragment -> 2 v_wmma per K-step).
// Double-buffered LDS staging, one barrier per K-step. Each staging thread
// owns a contiguous K-run: batched (clauseable) global loads, then a single
// packed 16B ds_store_b128 per run.
//   lA[2][32][40] row-major  (row m: 32 K values, pad 8, 16B-aligned runs)
//   lB[2][64][40] col-major  (col n: 32 K values, pad 8, 16B-aligned runs)
// A frag (lane lo,hi): K = (j>>3)*16 + hi*8 + (j&7) -> lA[.][m][hi*8], [m][16+hi*8]
// B frag (lane lo,hi): K = (j>>3)*16 + hi*8 + (j&7) -> lB[.][cn][hi*8], [cn][16+hi*8]
// ---------------------------------------------------------------------------
__global__ void gemm_wmma_kernel(const float* __restrict__ A, const float* __restrict__ B,
                                 const float* __restrict__ bias, float* __restrict__ C,
                                 int M, int N, int K, int lda, int ldb, int ldc, int act) {
  __shared__ __align__(16) _Float16 lA[2][32][40];
  __shared__ __align__(16) _Float16 lB[2][64][40];
  int tid  = threadIdx.x;
  int lane = tid & 31;
  int wave = tid >> 5;
  int m0 = blockIdx.y << 5;
  int n0 = blockIdx.x << 6;
  int lo = lane & 15;
  int hi = lane >> 4;
  int cn = (wave << 4) + lo;        // this wave's B column in LDS
  // staging ownership: contiguous K-runs per thread
  int ar = tid >> 2, akq = tid & 3;        // A: row 0..31, K-octet 0..3
  int bn = tid & 63, bkh = tid >> 6;       // B: col 0..63, K-half 0..1
  v8f acc0 = {}, acc1 = {};

  auto stageA = [&](int buf, int k0) {
    int gr = m0 + ar;
    float va[8];
    #pragma unroll
    for (int j = 0; j < 8; ++j) {
      int gk = k0 + (akq << 3) + j;
      bool ok = (gr < M) & (gk < K);
      size_t idx = ok ? ((size_t)gr * lda + gk) : 0;
      float v = A[idx];
      va[j] = ok ? v : 0.f;
    }
    v8h pk;
    #pragma unroll
    for (int j = 0; j < 8; ++j) pk[j] = (_Float16)va[j];
    *(v8h*)&lA[buf][ar][akq << 3] = pk;
  };
  auto stageB = [&](int buf, int k0) {
    int gn = n0 + bn;
    float vb[16];
    #pragma unroll
    for (int j = 0; j < 16; ++j) {
      int gk = k0 + (bkh << 4) + j;
      bool ok = (gk < K) & (gn < N);
      size_t idx = ok ? ((size_t)gk * ldb + gn) : 0;
      float v = B[idx];
      vb[j] = ok ? v : 0.f;
    }
    #pragma unroll
    for (int h = 0; h < 2; ++h) {
      v8h pk;
      #pragma unroll
      for (int j = 0; j < 8; ++j) pk[j] = (_Float16)vb[(h << 3) + j];
      *(v8h*)&lB[buf][bn][(bkh << 4) + (h << 3)] = pk;
    }
  };

  stageA(0, 0);
  stageB(0, 0);
  int buf = 0;
  for (int k0 = 0; k0 < K; k0 += 32) {
    __syncthreads();
    const v8h* pa0 = (const v8h*)&lA[buf][lo][hi << 3];
    const v8h* pa1 = (const v8h*)&lA[buf][lo][16 + (hi << 3)];
    v16h a0 = SHUF16(pa0[0], pa1[0]);
    const v8h* pa2 = (const v8h*)&lA[buf][16 + lo][hi << 3];
    const v8h* pa3 = (const v8h*)&lA[buf][16 + lo][16 + (hi << 3)];
    v16h a1 = SHUF16(pa2[0], pa3[0]);
    const v8h* pb = (const v8h*)&lB[buf][cn][hi << 3];
    v16h b = SHUF16(pb[0], pb[2]);
    if (k0 + 32 < K) {              // overlap next tile's loads with WMMAs
      stageA(buf ^ 1, k0 + 32);
      stageB(buf ^ 1, k0 + 32);
    }
    acc0 = __builtin_amdgcn_wmma_f32_16x16x32_f16(false, a0, false, b,
                                                  (short)0, acc0, false, false);
    acc1 = __builtin_amdgcn_wmma_f32_16x16x32_f16(false, a1, false, b,
                                                  (short)0, acc1, false, false);
    buf ^= 1;
  }
  int n = n0 + (wave << 4) + lo;
  if (n < N) {
    float bv = bias ? bias[n] : 0.f;
    #pragma unroll
    for (int r = 0; r < 8; ++r) {
      int m = m0 + (hi << 3) + r;
      if (m < M) {
        float v = acc0[r] + bv;
        if (act) v = v > 0.f ? v : 0.f;
        C[(size_t)m * ldc + n] = v;
      }
      int m2 = m + 16;
      if (m2 < M) {
        float v = acc1[r] + bv;
        if (act) v = v > 0.f ? v : 0.f;
        C[(size_t)m2 * ldc + n] = v;
      }
    }
  }
}

// ------------------------------ elementwise --------------------------------
__global__ void fill0_kernel(float* p, long long n) {
  long long i = (long long)blockIdx.x * blockDim.x + threadIdx.x;
  if (i < n) p[i] = 0.f;
}
__global__ void relu_kernel(float* p, long long n) {
  long long i = (long long)blockIdx.x * blockDim.x + threadIdx.x;
  if (i < n) p[i] = fmaxf(p[i], 0.f);
}
__global__ void bn_kernel(float* p, const float* g, const float* b,
                          const float* rm, const float* rv, long long n, int F) {
  long long i = (long long)blockIdx.x * blockDim.x + threadIdx.x;
  if (i >= n) return;
  int f = (int)(i % F);
  p[i] = (p[i] - rm[f]) * rsqrtf(rv[f] + 1e-5f) * g[f] + b[f];
}

// ------------------------------ LayerNorm ----------------------------------
__global__ void ln_kernel(const float* __restrict__ a, const float* __restrict__ res,
                          const float* __restrict__ g, const float* __restrict__ b,
                          float* __restrict__ out, int W) {
  __shared__ float sh[128];
  int row = blockIdx.x;
  const float* ap = a + (size_t)row * W;
  const float* rp = res ? res + (size_t)row * W : nullptr;
  float s = 0.f;
  for (int j = threadIdx.x; j < W; j += 128) s += ap[j] + (rp ? rp[j] : 0.f);
  sh[threadIdx.x] = s; __syncthreads();
  for (int o = 64; o > 0; o >>= 1) { if (threadIdx.x < o) sh[threadIdx.x] += sh[threadIdx.x + o]; __syncthreads(); }
  float mean = sh[0] / W; __syncthreads();
  float v = 0.f;
  for (int j = threadIdx.x; j < W; j += 128) { float t = ap[j] + (rp ? rp[j] : 0.f) - mean; v += t * t; }
  sh[threadIdx.x] = v; __syncthreads();
  for (int o = 64; o > 0; o >>= 1) { if (threadIdx.x < o) sh[threadIdx.x] += sh[threadIdx.x + o]; __syncthreads(); }
  float rstd = rsqrtf(sh[0] / W + 1e-5f);
  for (int j = threadIdx.x; j < W; j += 128) {
    float t = ap[j] + (rp ? rp[j] : 0.f);
    out[(size_t)row * W + j] = (t - mean) * rstd * g[j] + b[j];
  }
}

// ------------------------------ Mamba pieces -------------------------------
__global__ void conv_silu_kernel(const float* __restrict__ xz, const float* __restrict__ cw,
                                 const float* __restrict__ cb, float* __restrict__ out, int N) {
  long long i = (long long)blockIdx.x * blockDim.x + threadIdx.x;
  if (i >= (long long)N * DI_) return;
  int n = (int)(i / DI_), c = (int)(i % DI_);
  float s = cb[c];
  #pragma unroll
  for (int k = 0; k < DC_; ++k) {
    int t = n - (DC_ - 1) + k;
    if (t >= 0) s += cw[c * DC_ + k] * xz[(size_t)t * (2 * DI_) + c];
  }
  out[i] = s / (1.f + __expf(-s));  // silu
}

__global__ void dt_kernel(const float* __restrict__ dbc, const float* __restrict__ dt_w,
                          const float* __restrict__ dt_b, float* __restrict__ dt, int N) {
  long long i = (long long)blockIdx.x * blockDim.x + threadIdx.x;
  if (i >= (long long)N * DI_) return;
  int n = (int)(i / DI_), c = (int)(i % DI_);
  float s = dt_b[c];
  #pragma unroll
  for (int t = 0; t < DTR_; ++t) s += dbc[(size_t)n * DBC_ + t] * dt_w[t * DI_ + c];
  dt[i] = (s > 20.f) ? s : log1pf(__expf(s));  // softplus
}

// Sequential selective scan: one thread per channel, 32-wide state in regs.
__global__ void scan_kernel(const float* __restrict__ dt, const float* __restrict__ dbc,
                            const float* __restrict__ xsa, const float* __restrict__ A_log,
                            float* __restrict__ ys, int N) {
  int c = threadIdx.x;
  if (c >= DI_) return;
  float A[DS_], h[DS_];
  #pragma unroll
  for (int s = 0; s < DS_; ++s) { A[s] = -__expf(A_log[c * DS_ + s]); h[s] = 0.f; }
  for (int n = 0; n < N; ++n) {
    float d = dt[(size_t)n * DI_ + c];
    float x = xsa[(size_t)n * DI_ + c];
    const float* Bp = dbc + (size_t)n * DBC_ + DTR_;
    const float* Cp = Bp + DS_;
    float y = 0.f;
    #pragma unroll
    for (int s = 0; s < DS_; ++s) {
      h[s] = __expf(d * A[s]) * h[s] + d * x * Bp[s];
      y += h[s] * Cp[s];
    }
    ys[(size_t)n * DI_ + c] = y;
  }
}

__global__ void gate_kernel(const float* __restrict__ ys, const float* __restrict__ xsa,
                            const float* __restrict__ Dp, const float* __restrict__ xz,
                            float* __restrict__ yg, int N) {
  long long i = (long long)blockIdx.x * blockDim.x + threadIdx.x;
  if (i >= (long long)N * DI_) return;
  int n = (int)(i / DI_), c = (int)(i % DI_);
  float z = xz[(size_t)n * (2 * DI_) + DI_ + c];
  float sz = z / (1.f + __expf(-z));
  yg[i] = (ys[i] + xsa[i] * Dp[c]) * sz;
}

// ------------------------------ GCN pieces ---------------------------------
__global__ void deg_count_kernel(const int* __restrict__ dst, float* __restrict__ deg, int E) {
  int i = blockIdx.x * blockDim.x + threadIdx.x;
  if (i < E) atomicAdd(&deg[dst[i]], 1.f);
}
__global__ void dinv_kernel(const float* __restrict__ deg, float* __restrict__ dinv, int n) {
  int i = blockIdx.x * blockDim.x + threadIdx.x;
  if (i < n) dinv[i] = rsqrtf(deg[i] + 1.f);   // +1 self-loop
}
__global__ void coef_kernel(const int* __restrict__ src, const int* __restrict__ dst,
                            const float* __restrict__ dinv, float* __restrict__ coef, int E) {
  int i = blockIdx.x * blockDim.x + threadIdx.x;
  if (i < E) coef[i] = dinv[src[i]] * dinv[dst[i]];
}
__global__ void gcn_self_bias_kernel(const float* __restrict__ h, const float* __restrict__ dinv,
                                     const float* __restrict__ b, float* __restrict__ out,
                                     long long n, int F) {
  long long i = (long long)blockIdx.x * blockDim.x + threadIdx.x;
  if (i >= n) return;
  int node = (int)(i / F), f = (int)(i % F);
  float di = dinv[node];
  out[i] = h[i] * di * di + b[f];
}
__global__ void gcn_scatter_kernel(const float* __restrict__ h, const int* __restrict__ src,
                                   const int* __restrict__ dst, const float* __restrict__ coef,
                                   float* __restrict__ out, long long E, int F) {
  long long i = (long long)blockIdx.x * blockDim.x + threadIdx.x;
  if (i >= E * F) return;
  long long e = i / F; int f = (int)(i % F);
  atomicAdd(&out[(size_t)dst[e] * F + f], h[(size_t)src[e] * F + f] * coef[e]);
}

// ------------------------------ pooling / gather ---------------------------
__global__ void pool_sum_kernel(const float* __restrict__ x, const int* __restrict__ seg,
                                float* __restrict__ sum, long long n, int F) {
  long long i = (long long)blockIdx.x * blockDim.x + threadIdx.x;
  if (i >= n) return;
  int node = (int)(i / F), f = (int)(i % F);
  atomicAdd(&sum[(size_t)seg[node] * F + f], x[i]);
}
__global__ void pool_cnt_kernel(const int* __restrict__ seg, float* __restrict__ cnt, int n) {
  int i = blockIdx.x * blockDim.x + threadIdx.x;
  if (i < n) atomicAdd(&cnt[seg[i]], 1.f);
}
__global__ void pool_div_kernel(float* __restrict__ sum, const float* __restrict__ cnt,
                                long long n, int F) {
  long long i = (long long)blockIdx.x * blockDim.x + threadIdx.x;
  if (i < n) sum[i] = sum[i] / cnt[i / F];
}
__global__ void gather_kernel(const float* __restrict__ in, const int* __restrict__ idx,
                              float* __restrict__ out, int R, int F) {
  int i = blockIdx.x * blockDim.x + threadIdx.x;
  if (i >= R * F) return;
  int r = i / F, f = i % F;
  out[i] = in[(size_t)idx[r] * F + f];
}

// ------------------------------ host helpers -------------------------------
static inline unsigned gb(long long n) { return (unsigned)((n + 255) / 256); }

static void gemm(hipStream_t s, const float* A, const float* B, const float* bias,
                 float* C, int M, int N, int K, int lda, int ldb, int ldc, int act) {
  dim3 grid((N + 63) / 64, (M + 31) / 32);
  gemm_wmma_kernel<<<grid, dim3(128), 0, s>>>(A, B, bias, C, M, N, K, lda, ldb, ldc, act);
}

static void gcn_layer(hipStream_t s, const float* x, const float* W, const float* b,
                      const int* src, const int* dst, const float* dinv, const float* coef,
                      float* h, float* out, int n, long long E, int Fin, int Fout) {
  gemm(s, x, W, nullptr, h, n, Fout, Fin, Fin, Fout, Fout, 0);
  long long tot = (long long)n * Fout;
  gcn_self_bias_kernel<<<gb(tot), 256, 0, s>>>(h, dinv, b, out, tot, Fout);
  long long etot = E * (long long)Fout;
  gcn_scatter_kernel<<<gb(etot), 256, 0, s>>>(h, src, dst, coef, out, E, Fout);
  relu_kernel<<<gb(tot), 256, 0, s>>>(out, tot);
}

// ---------------------------------------------------------------------------
extern "C" void kernel_launch(void* const* d_in, const int* in_sizes, int n_in,
                              void* d_out, int out_size, void* d_ws, size_t ws_size,
                              hipStream_t stream) {
  (void)in_sizes; (void)n_in; (void)out_size; (void)ws_size;
  #define PF(i) ((const float*)d_in[(i)])
  #define PI(i) ((const int*)d_in[(i)])
  const float* mol_x   = PF(0);
  const float* pro_x   = PF(1);
  const int* mol_src   = PI(2);
  const int* mol_dst   = PI(3);
  const int* mol_batch = PI(4);
  const int* pro_src   = PI(5);
  const int* pro_dst   = PI(6);
  const int* pro_batch = PI(7);
  const int* ppi_src   = PI(8);
  const int* ppi_dst   = PI(9);
  const int* seq_num   = PI(10);
  const int P = 11;                       // params, insertion order; mamba = 3 x 13 leaves
  const int G = P + 39;                   // after mamba layers

  // ---- workspace bump allocation (~335 MB) ----
  char* wp = (char*)d_ws;
  auto alloc = [&](size_t elems) -> float* {
    float* p = (float*)wp;
    wp += ((elems * sizeof(float)) + 255) & ~(size_t)255;
    return p;
  };
  float* R1 = alloc((size_t)NMOL * 312);             // xz / GCN h
  float* R2 = alloc((size_t)NMOL * 312);             // xsa|ys ; later GCN ping
  float* R3 = alloc((size_t)NMOL * 312);             // dt|yg ; later GCN pong
  float* R4 = alloc((size_t)NMOL * (DBC_ + 2 * D_)); // dbc|ymid|yp
  float* xA = alloc((size_t)NMOL * D_);
  float* xB = alloc((size_t)NMOL * D_);
  float* molDeg  = alloc(NMOL);
  float* molDinv = alloc(NMOL);
  float* molCoef = alloc(EMOL);
  float* molPool = alloc((size_t)BMOL * OD_);
  float* molCnt  = alloc(BMOL);
  float* molFC   = alloc((size_t)BMOL * 1024);
  float* molEmb  = alloc((size_t)BMOL * OD_);
  float* proA = alloc((size_t)NPRO * OD_);
  float* proB = alloc((size_t)NPRO * OD_);
  float* proDeg  = alloc(NPRO);
  float* proDinv = alloc(NPRO);
  float* proCoef = alloc(EPRO);
  float* proPool = alloc((size_t)NPROT * OD_);
  float* proCnt  = alloc(NPROT);
  float* proFC   = alloc((size_t)NPROT * 1024);
  float* proEmb  = alloc((size_t)NPROT * OD_);
  float* ppiDeg  = alloc(NPROT);
  float* ppiDinv = alloc(NPROT);
  float* ppiCoef = alloc(EPPI);
  float* ppiH = alloc((size_t)NPROT * 1024);
  float* ppiO = alloc((size_t)NPROT * 1024);
  float* pp1  = alloc((size_t)NPROT * OD_);
  float* pp2  = alloc((size_t)NPROT * OD_);
  float* ppg  = alloc((size_t)BMOL * OD_);
  float* hv   = alloc((size_t)BMOL * OD_);
  float* c1   = alloc((size_t)BMOL * OD_);
  float* c2   = alloc((size_t)BMOL * OD_);
  float* c3   = alloc((size_t)BMOL * OD_);
  float* c4   = alloc((size_t)BMOL * OD_);
  float* hHd  = alloc((size_t)BMOL * 1024);
  float* h2Hd = alloc((size_t)BMOL * 512);

  float* xsa  = R2;
  float* ysB  = R2 + (size_t)NMOL * DI_;
  float* dtB  = R3;
  float* ygB  = R3 + (size_t)NMOL * DI_;
  float* dbcB = R4;
  float* ymid = R4 + (size_t)NMOL * DBC_;
  float* yp   = ymid + (size_t)NMOL * D_;

  // ================= Phase 1: 3 x Mamba + proj + LN ==================
  const float* xin = mol_x;
  for (int L = 0; L < 3; ++L) {
    int pb = P + L * 13;
    const float* in_proj  = PF(pb + 0);   // [78,312]
    const float* conv_w   = PF(pb + 1);   // [156,4]
    const float* conv_b   = PF(pb + 2);
    const float* x_proj   = PF(pb + 3);   // [156,69]
    const float* dt_w     = PF(pb + 4);   // [5,156]
    const float* dt_b     = PF(pb + 5);
    const float* A_log    = PF(pb + 6);   // [156,32]
    const float* Dp       = PF(pb + 7);
    const float* out_proj = PF(pb + 8);   // [156,78]
    const float* proj_w   = PF(pb + 9);   // [78,78]
    const float* proj_b   = PF(pb + 10);
    const float* ln_g     = PF(pb + 11);
    const float* ln_b     = PF(pb + 12);

    gemm(stream, xin, in_proj, nullptr, R1, NMOL, 2 * DI_, D_, D_, 2 * DI_, 2 * DI_, 0);
    conv_silu_kernel<<<gb((long long)NMOL * DI_), 256, 0, stream>>>(R1, conv_w, conv_b, xsa, NMOL);
    gemm(stream, xsa, x_proj, nullptr, dbcB, NMOL, DBC_, DI_, DI_, DBC_, DBC_, 0);
    dt_kernel<<<gb((long long)NMOL * DI_), 256, 0, stream>>>(dbcB, dt_w, dt_b, dtB, NMOL);
    scan_kernel<<<1, 192, 0, stream>>>(dtB, dbcB, xsa, A_log, ysB, NMOL);
    gate_kernel<<<gb((long long)NMOL * DI_), 256, 0, stream>>>(ysB, xsa, Dp, R1, ygB, NMOL);
    gemm(stream, ygB, out_proj, nullptr, ymid, NMOL, D_, DI_, DI_, D_, D_, 0);
    gemm(stream, ymid, proj_w, proj_b, yp, NMOL, D_, D_, D_, D_, D_, 0);
    float* xout = (L == 1) ? xB : xA;     // L0->xA, L1->xB, L2->xA
    ln_kernel<<<NMOL, 128, 0, stream>>>(yp, xin, ln_g, ln_b, xout, D_);
    xin = xout;
  }

  // ================= Phase 2: molecule GCN + pool + FC ===============
  fill0_kernel<<<gb(NMOL), 256, 0, stream>>>(molDeg, NMOL);
  deg_count_kernel<<<gb(EMOL), 256, 0, stream>>>(mol_dst, molDeg, EMOL);
  dinv_kernel<<<gb(NMOL), 256, 0, stream>>>(molDeg, molDinv, NMOL);
  coef_kernel<<<gb(EMOL), 256, 0, stream>>>(mol_src, mol_dst, molDinv, molCoef, EMOL);

  gcn_layer(stream, xin, PF(G + 0), PF(G + 1), mol_src, mol_dst, molDinv, molCoef,
            R1, R2, NMOL, EMOL, D_, 2 * D_);
  gcn_layer(stream, R2, PF(G + 2), PF(G + 3), mol_src, mol_dst, molDinv, molCoef,
            R1, R3, NMOL, EMOL, 2 * D_, 4 * D_);
  gcn_layer(stream, R3, PF(G + 4), PF(G + 5), mol_src, mol_dst, molDinv, molCoef,
            R1, R2, NMOL, EMOL, 4 * D_, OD_);

  fill0_kernel<<<gb((long long)BMOL * OD_), 256, 0, stream>>>(molPool, (long long)BMOL * OD_);
  fill0_kernel<<<gb(BMOL), 256, 0, stream>>>(molCnt, BMOL);
  pool_sum_kernel<<<gb((long long)NMOL * OD_), 256, 0, stream>>>(R2, mol_batch, molPool, (long long)NMOL * OD_, OD_);
  pool_cnt_kernel<<<gb(NMOL), 256, 0, stream>>>(mol_batch, molCnt, NMOL);
  pool_div_kernel<<<gb((long long)BMOL * OD_), 256, 0, stream>>>(molPool, molCnt, (long long)BMOL * OD_, OD_);

  gemm(stream, molPool, PF(G + 6), PF(G + 7), molFC, BMOL, 1024, OD_, OD_, 1024, 1024, 1);
  gemm(stream, molFC, PF(G + 8), PF(G + 9), molEmb, BMOL, OD_, 1024, 1024, OD_, OD_, 0);

  // ================= Phase 3: protein GCN + pool + FC ================
  fill0_kernel<<<gb(NPRO), 256, 0, stream>>>(proDeg, NPRO);
  deg_count_kernel<<<gb(EPRO), 256, 0, stream>>>(pro_dst, proDeg, EPRO);
  dinv_kernel<<<gb(NPRO), 256, 0, stream>>>(proDeg, proDinv, NPRO);
  coef_kernel<<<gb(EPRO), 256, 0, stream>>>(pro_src, pro_dst, proDinv, proCoef, EPRO);

  gcn_layer(stream, pro_x, PF(G + 10), PF(G + 11), pro_src, pro_dst, proDinv, proCoef,
            proA, proB, NPRO, EPRO, 33, OD_);
  bn_kernel<<<gb((long long)NPRO * OD_), 256, 0, stream>>>(proB, PF(G + 20), PF(G + 21), PF(G + 22), PF(G + 23), (long long)NPRO * OD_, OD_);
  gcn_layer(stream, proB, PF(G + 12), PF(G + 13), pro_src, pro_dst, proDinv, proCoef,
            proA, proB, NPRO, EPRO, OD_, OD_);
  bn_kernel<<<gb((long long)NPRO * OD_), 256, 0, stream>>>(proB, PF(G + 24), PF(G + 25), PF(G + 26), PF(G + 27), (long long)NPRO * OD_, OD_);
  gcn_layer(stream, proB, PF(G + 14), PF(G + 15), pro_src, pro_dst, proDinv, proCoef,
            proA, proB, NPRO, EPRO, OD_, OD_);
  bn_kernel<<<gb((long long)NPRO * OD_), 256, 0, stream>>>(proB, PF(G + 28), PF(G + 29), PF(G + 30), PF(G + 31), (long long)NPRO * OD_, OD_);

  fill0_kernel<<<gb((long long)NPROT * OD_), 256, 0, stream>>>(proPool, (long long)NPROT * OD_);
  fill0_kernel<<<gb(NPROT), 256, 0, stream>>>(proCnt, NPROT);
  pool_sum_kernel<<<gb((long long)NPRO * OD_), 256, 0, stream>>>(proB, pro_batch, proPool, (long long)NPRO * OD_, OD_);
  pool_cnt_kernel<<<gb(NPRO), 256, 0, stream>>>(pro_batch, proCnt, NPRO);
  pool_div_kernel<<<gb((long long)NPROT * OD_), 256, 0, stream>>>(proPool, proCnt, (long long)NPROT * OD_, OD_);

  gemm(stream, proPool, PF(G + 16), PF(G + 17), proFC, NPROT, 1024, OD_, OD_, 1024, 1024, 1);
  gemm(stream, proFC, PF(G + 18), PF(G + 19), proEmb, NPROT, OD_, 1024, 1024, OD_, OD_, 0);

  // ================= Phase 4: PPI GCN + FC ===========================
  fill0_kernel<<<gb(NPROT), 256, 0, stream>>>(ppiDeg, NPROT);
  deg_count_kernel<<<gb(EPPI), 256, 0, stream>>>(ppi_dst, ppiDeg, EPPI);
  dinv_kernel<<<gb(NPROT), 256, 0, stream>>>(ppiDeg, ppiDinv, NPROT);
  coef_kernel<<<gb(EPPI), 256, 0, stream>>>(ppi_src, ppi_dst, ppiDinv, ppiCoef, EPPI);

  gcn_layer(stream, proEmb, PF(G + 32), PF(G + 33), ppi_src, ppi_dst, ppiDinv, ppiCoef,
            ppiH, ppiO, NPROT, EPPI, OD_, 1024);
  gcn_layer(stream, ppiO, PF(G + 34), PF(G + 35), ppi_src, ppi_dst, ppiDinv, ppiCoef,
            pp1, pp2, NPROT, EPPI, 1024, OD_);
  gemm(stream, pp2, PF(G + 36), PF(G + 37), ppiH, NPROT, 1024, OD_, OD_, 1024, 1024, 1);
  gemm(stream, ppiH, PF(G + 38), PF(G + 39), pp1, NPROT, OD_, 1024, 1024, OD_, OD_, 0);

  // ================= Phase 5: head ===================================
  gather_kernel<<<gb((long long)BMOL * OD_), 256, 0, stream>>>(pp1, seq_num, ppg, BMOL, OD_);
  // v = pp @ Wv + bv  (Wv = attn_in_w[:, 2*OD:], ldb = 3*OD)
  gemm(stream, ppg, PF(G + 40) + 2 * OD_, PF(G + 41) + 2 * OD_, hv, BMOL, OD_, OD_, OD_, 3 * OD_, OD_, 0);
  gemm(stream, hv, PF(G + 42), PF(G + 43), c1, BMOL, OD_, OD_, OD_, OD_, OD_, 0);   // attn_out
  gemm(stream, c1, PF(G + 44), PF(G + 45), c2, BMOL, OD_, OD_, OD_, OD_, OD_, 0);   // cap
  ln_kernel<<<BMOL, 128, 0, stream>>>(c2, molEmb, PF(G + 46), PF(G + 47), c3, OD_);
  ln_kernel<<<BMOL, 128, 0, stream>>>(c3, ppg, PF(G + 46), PF(G + 47), c4, OD_);
  gemm(stream, c4, PF(G + 48), PF(G + 49), hHd, BMOL, 1024, OD_, OD_, 1024, 1024, 1);
  gemm(stream, hHd, PF(G + 50), PF(G + 51), h2Hd, BMOL, 512, 1024, 1024, 512, 512, 1);
  gemm(stream, h2Hd, PF(G + 52), PF(G + 53), (float*)d_out, BMOL, 1, 512, 512, 1, 1, 0);
  #undef PF
  #undef PI
}
